// netVLADlayer_48189533061458
// MI455X (gfx1250) — matrix operand order repaired and checked by reading, not compiled, not run on hardware
//
#include <hip/hip_runtime.h>
#include <hip/hip_bf16.h>
#include <cstdint>
#include <cstddef>

// ---------------------------------------------------------------------------
// NetVLAD on gfx1250 (MI455X), wave32 + WMMA bf16.
//
// Phase 1 (dominant): f = x@W + b  (M=65536,K=512,N=64) -> 4.3 GF, 134 MB read.
//   AI = 32 FLOP/B -> memory-bound only if matrix throughput >= ~740 TF,
//   so we use v_wmma_f32_16x16x32_bf16 (f32 accumulate), not the K=4 f32 WMMA.
// Phase 2: V'[k,d] = sum_p s[p,k] f[p,d] + C[d,k]*colsum[k]  (bf16 WMMA,
//   operands stored transposed by phase 1 so every lane load is contiguous).
// Phase 3: per-cluster + global L2 normalization.
// ---------------------------------------------------------------------------

typedef __attribute__((ext_vector_type(16))) __bf16 v16bf;
typedef __attribute__((ext_vector_type(8)))  __bf16 v8bf;
typedef __attribute__((ext_vector_type(8)))  float  v8f;
typedef __attribute__((ext_vector_type(4)))  float  v4f;

#define EPSN 1e-12f

// --------------------------------------------------------------------------
// k0: W [512][64] f32  ->  Wp [64][512] bf16  (column-major W = rows of W^T,
// so phase-1 B-fragments are contiguous 32B per lane)
// --------------------------------------------------------------------------
__global__ void k0_packW(const float* __restrict__ W, __bf16* __restrict__ Wp) {
  int i = blockIdx.x * blockDim.x + threadIdx.x;   // 0 .. 64*512-1
  if (i >= 64 * 512) return;
  int k = i >> 9;
  int c = i & 511;
  Wp[i] = (__bf16)W[c * 64 + k];
}

// --------------------------------------------------------------------------
// k1: GEMM (x@W+b) + softmax, write fT[b][64][4096] and sT[b][64][4096] bf16.
// One wave = 16 pixels x 64 clusters (4 accumulator tiles).
// A-frag (16x32 bf16): lane<16 holds K {c..c+15}, lane>=16 K {c+16..c+31}.
// B-frag: lane's column n = nt*16+(lane&15), 16 contiguous
// K values from Wp row n.  C/D layout: (vgpr r, lane) = (M=r+8*half, N=lane&15).
// --------------------------------------------------------------------------
__global__ void __launch_bounds__(256) k1_gemm_softmax(
    const float* __restrict__ x, const __bf16* __restrict__ Wp,
    const float* __restrict__ bias,
    __bf16* __restrict__ fT, __bf16* __restrict__ sT) {
  const int lane = threadIdx.x & 31;
  const int wave = threadIdx.x >> 5;
  const int l15  = lane & 15;
  const int half = lane >> 4;
  const int pixbase = (blockIdx.x * 8 + wave) * 16;   // global pixel tile
  const int b  = pixbase >> 12;                        // batch (4096 px/batch)
  const int pl = pixbase & 4095;                       // pixel within batch

  v8f acc[4];
#pragma unroll
  for (int nt = 0; nt < 4; ++nt) acc[nt] = (v8f)(0.0f);

  const float* xrow = x + (size_t)(pixbase + l15) * 512;

  for (int cb = 0; cb < 512; cb += 32) {
    const float* xp = xrow + cb + half * 16;
    v4f x0 = *(const v4f*)(xp);
    v4f x1 = *(const v4f*)(xp + 4);
    v4f x2 = *(const v4f*)(xp + 8);
    v4f x3 = *(const v4f*)(xp + 12);
    v16bf a;
#pragma unroll
    for (int j = 0; j < 4; ++j) {
      a[j]      = (__bf16)x0[j];
      a[4 + j]  = (__bf16)x1[j];
      a[8 + j]  = (__bf16)x2[j];
      a[12 + j] = (__bf16)x3[j];
    }
#pragma unroll
    for (int nt = 0; nt < 4; ++nt) {
      const __bf16* wp = Wp + (size_t)(nt * 16 + l15) * 512 + cb + half * 16;
      v16bf bf = *(const v16bf*)wp;                    // 32B contiguous
      acc[nt] = __builtin_amdgcn_wmma_f32_16x16x32_bf16(
          false, a, false, bf, (short)0, acc[nt], false, false);
    }
  }

  // bias (same for all 8 rows held by this lane in a given N-tile)
#pragma unroll
  for (int nt = 0; nt < 4; ++nt) {
    float bv = bias[nt * 16 + l15];
#pragma unroll
    for (int r = 0; r < 8; ++r) acc[nt][r] += bv;
  }

  // Softmax over the 64 clusters of each pixel row. Row M = r + 8*half lives
  // in one 16-lane half, across the 4 N-tiles -> xor-shuffles 1,2,4,8 stay
  // inside the half.
  v8f sm[4];
#pragma unroll
  for (int r = 0; r < 8; ++r) {
    float f0 = acc[0][r], f1 = acc[1][r], f2 = acc[2][r], f3 = acc[3][r];
    float mx = fmaxf(fmaxf(f0, f1), fmaxf(f2, f3));
#pragma unroll
    for (int d = 1; d < 16; d <<= 1) mx = fmaxf(mx, __shfl_xor(mx, d, 32));
    float e0 = __expf(f0 - mx), e1 = __expf(f1 - mx);
    float e2 = __expf(f2 - mx), e3 = __expf(f3 - mx);
    float sum = e0 + e1 + e2 + e3;
#pragma unroll
    for (int d = 1; d < 16; d <<= 1) sum += __shfl_xor(sum, d, 32);
    float inv = 1.0f / sum;
    sm[0][r] = e0 * inv; sm[1][r] = e1 * inv;
    sm[2][r] = e2 * inv; sm[3][r] = e3 * inv;
  }

  // Store transposed (index-major over p): per lane, 8 consecutive pixels
  // (r = 0..7 at p = pl + 8*half + r) for column n -> one 16B store each.
#pragma unroll
  for (int nt = 0; nt < 4; ++nt) {
    int n = nt * 16 + l15;
    size_t base = ((size_t)b * 64 + n) * 4096 + pl + half * 8;
    v8bf pf, ps;
#pragma unroll
    for (int r = 0; r < 8; ++r) {
      pf[r] = (__bf16)acc[nt][r];
      ps[r] = (__bf16)sm[nt][r];
    }
    *(v8bf*)(fT + base) = pf;   // logits (with bias), transposed
    *(v8bf*)(sT + base) = ps;   // softmax, transposed
  }
}

// --------------------------------------------------------------------------
// k2: per (batch, 512-pixel slice): D[k,d] = sum_p sT[k][p] * fT[d][p]
// (= V' directly in output orientation). A = sT rows, B^T = fT rows -> both
// per-lane contiguous. Partials to ws; colsum via per-wave reduction of the
// A-fragment values (8192 atomics total).
// --------------------------------------------------------------------------
__global__ void __launch_bounds__(128) k2_vlad_gemm(
    const __bf16* __restrict__ sT, const __bf16* __restrict__ fT,
    float* __restrict__ part, float* __restrict__ colsum) {
  const int lane  = threadIdx.x & 31;
  const int ktile = threadIdx.x >> 5;   // 0..3
  const int l15   = lane & 15;
  const int half  = lane >> 4;
  const int slice = blockIdx.x;         // 0..7
  const int b     = blockIdx.y;         // 0..15

  v8f acc[4];
#pragma unroll
  for (int dt = 0; dt < 4; ++dt) acc[dt] = (v8f)(0.0f);
  float cs = 0.0f;

  const __bf16* arow = sT + ((size_t)b * 64 + ktile * 16 + l15) * 4096;
  const int p0 = slice * 512;

  for (int pc = p0; pc < p0 + 512; pc += 32) {
    v8bf a0 = *(const v8bf*)(arow + pc + half * 16);
    v8bf a1 = *(const v8bf*)(arow + pc + half * 16 + 8);
    v16bf a;
#pragma unroll
    for (int j = 0; j < 8; ++j) { a[j] = a0[j]; a[8 + j] = a1[j]; }
#pragma unroll
    for (int j = 0; j < 8; ++j) cs += (float)a0[j] + (float)a1[j];
#pragma unroll
    for (int dt = 0; dt < 4; ++dt) {
      const __bf16* bp =
          fT + ((size_t)b * 64 + dt * 16 + l15) * 4096 + pc + half * 16;
      v16bf bf = *(const v16bf*)bp;                   // 32B contiguous
      acc[dt] = __builtin_amdgcn_wmma_f32_16x16x32_bf16(
          false, a, false, bf, (short)0, acc[dt], false, false);
    }
  }

  // partials: part[b][slice][k][d] f32, D element (r,lane) = (M=r+8*half, N=l15)
#pragma unroll
  for (int dt = 0; dt < 4; ++dt)
#pragma unroll
    for (int r = 0; r < 8; ++r)
      part[((size_t)(b * 8 + slice) * 64 + ktile * 16 + half * 8 + r) * 64 +
           dt * 16 + l15] = acc[dt][r];

  // colsum[k] = sum_p s[p,k]: lanes l15 and l15+16 hold disjoint p-halves of
  // the same row k -> combine across halves, one atomic per (k, wave, slice).
  cs += __shfl_xor(cs, 16, 32);
  if (half == 0)
    atomicAdd(colsum + b * 64 + ktile * 16 + l15, cs);
}

// --------------------------------------------------------------------------
// k3: per batch: reduce slice partials + C-term, per-cluster L2 normalize,
// global L2 normalize, write f32 output [b][k*64+d].
// --------------------------------------------------------------------------
__global__ void __launch_bounds__(256) k3_finalize(
    const float* __restrict__ part, const float* __restrict__ colsum,
    const float* __restrict__ C, float* __restrict__ out) {
  __shared__ float vbuf[64 * 65];   // +1 pad per row: stride 65 kills conflicts
  __shared__ float rinv[64];
  __shared__ float rowss[64];
  __shared__ float finv;
  const int b = blockIdx.x;
  const int t = threadIdx.x;

  for (int i = 0; i < 16; ++i) {
    int e = t + i * 256;            // 0..4095
    int k = e >> 6, d = e & 63;
    float v = 0.0f;
#pragma unroll
    for (int s = 0; s < 8; ++s)
      v += part[((size_t)(b * 8 + s) * 64 + k) * 64 + d];
    v += C[d * 64 + k] * colsum[b * 64 + k];   // C is [dim][clusters] = [d][k]
    vbuf[k * 65 + d] = v;
  }
  __syncthreads();

  if (t < 64) {
    float ss = 0.0f;
    for (int d = 0; d < 64; ++d) { float v = vbuf[t * 65 + d]; ss += v * v; }
    float iv = rsqrtf(fmaxf(ss, EPSN));
    rinv[t]  = iv;
    rowss[t] = ss * iv * iv;        // sum-sq of the normalized row
  }
  __syncthreads();
  if (t == 0) {
    float fs = 0.0f;
    for (int k = 0; k < 64; ++k) fs += rowss[k];
    finv = rsqrtf(fmaxf(fs, EPSN));
  }
  __syncthreads();

  for (int i = 0; i < 16; ++i) {
    int e = t + i * 256;
    int k = e >> 6, d = e & 63;
    out[(size_t)b * 4096 + e] = vbuf[k * 65 + d] * rinv[k] * finv;
  }
}

// --------------------------------------------------------------------------
// Launch. Workspace layout (bytes):
//   [0, 64K)         Wp   : 64*512 bf16
//   [64K, +8M)       fT   : 16*64*4096 bf16
//   [.., +8M)        sT   : 16*64*4096 bf16
//   [.., +2M)        part : 16*8*64*64 f32
//   [.., +4K)        colsum: 16*64 f32   (memset to 0 each call)
// Total ~18.9 MB.
// --------------------------------------------------------------------------
extern "C" void kernel_launch(void* const* d_in, const int* in_sizes, int n_in,
                              void* d_out, int out_size, void* d_ws,
                              size_t ws_size, hipStream_t stream) {
  const float* x    = (const float*)d_in[0];
  const float* W    = (const float*)d_in[1];
  const float* bias = (const float*)d_in[2];
  const float* C    = (const float*)d_in[3];
  float* out = (float*)d_out;

  char* ws = (char*)d_ws;
  const size_t OFF_WP   = 0;
  const size_t OFF_FT   = 65536;
  const size_t OFF_ST   = OFF_FT + (size_t)16 * 64 * 4096 * 2;
  const size_t OFF_PART = OFF_ST + (size_t)16 * 64 * 4096 * 2;
  const size_t OFF_CS   = OFF_PART + (size_t)16 * 8 * 64 * 64 * 4;

  __bf16* Wp    = (__bf16*)(ws + OFF_WP);
  __bf16* fT    = (__bf16*)(ws + OFF_FT);
  __bf16* sT    = (__bf16*)(ws + OFF_ST);
  float* part   = (float*)(ws + OFF_PART);
  float* colsum = (float*)(ws + OFF_CS);

  k0_packW<<<128, 256, 0, stream>>>(W, Wp);
  k1_gemm_softmax<<<512, 256, 0, stream>>>(x, Wp, bias, fT, sT);
  hipMemsetAsync(colsum, 0, 16 * 64 * sizeof(float), stream);
  dim3 g2(8, 16);
  k2_vlad_gemm<<<g2, 128, 0, stream>>>(sT, fT, part, colsum);
  k3_finalize<<<16, 256, 0, stream>>>(part, colsum, C, out);
}
